// Head_54820962566740
// MI455X (gfx1250) — matrix operand order repaired and checked by reading, not compiled, not run on hardware
//
#include <hip/hip_runtime.h>
#include <hip/hip_bf16.h>

// ---------------------------------------------------------------------------
// Single-head causal attention (B=8, T=2048, D=1024, H=128) for gfx1250.
// bf16 WMMA (v_wmma_f32_16x16x32_bf16) for QKV projection, QK^T and P*V.
// Flash-attention online softmax; V stored transposed; weights pre-packed
// into WMMA B-fragment order. All matrix math f32-accumulated.
// f32->bf16 via native cast so the backend can emit v_cvt_pk_bf16_f32.
// ---------------------------------------------------------------------------

#define NEMBD 1024
#define HS    128
#define TT    2048
#define BB    8

typedef __bf16 bf16_t;
typedef __bf16 v16bf __attribute__((ext_vector_type(16)));
typedef float  v8f   __attribute__((ext_vector_type(8)));

union Frag16 {
  v16bf v;
  uint4 q[2];
  unsigned short s[16];
  __bf16 h[16];
};

static __device__ __forceinline__ unsigned short f32_bf16_bits(float f) {
  // Native conversion (round-to-nearest-even); lowers to v_cvt_*bf16_f32 and
  // pairs of these get packed into v_cvt_pk_bf16_f32.
  return __builtin_bit_cast(unsigned short, (__bf16)f);
}

// ---------------------------------------------------------------------------
// Kernel 0: repack Wq/Wk/Wv (f32, 1024x128 row-major) into bf16 laid out in
// per-lane WMMA B-fragment order (k contiguous per lane half): [mat][kb(32)][n8(8)][lane(32)][e(16)].
// value(lane,e) = W[(kb*32 + (lane>>4)*16 + e)*128 + n8*16 + (lane&15)]
// ---------------------------------------------------------------------------
__global__ void pack_weights(const float* __restrict__ Wq,
                             const float* __restrict__ Wk,
                             const float* __restrict__ Wv,
                             unsigned short* __restrict__ wpack) {
  int id   = blockIdx.x * blockDim.x + threadIdx.x;  // 0..24575
  int lane = id & 31;
  int n8   = (id >> 5) & 7;
  int kb   = (id >> 8) & 31;
  int mat  = id >> 13;                               // 0..2
  const float* W = (mat == 0) ? Wq : (mat == 1) ? Wk : Wv;
  int idx = lane & 15, hl = lane >> 4;
  int col = n8 * 16 + idx;
  unsigned short* dst =
      wpack + ((size_t)((mat * 32 + kb) * 8 + n8) * 32 + lane) * 16;
#pragma unroll
  for (int e = 0; e < 16; ++e) {
    dst[e] = f32_bf16_bits(W[(kb * 32 + hl * 16 + e) * 128 + col]);
  }
}

// ---------------------------------------------------------------------------
// Kernel 1: QKV projection + RoPE.
// One wave computes a 16-row tile of one of {Q,K,V}: 8 N-tiles x 32 K-steps
// of v_wmma_f32_16x16x32_bf16. RoPE on Q/K via shfl_xor(…,1) pair mixing.
// Q,K stored row-major bf16 [B][T][128]; V stored transposed [B][128][T].
// ---------------------------------------------------------------------------
__global__ void __launch_bounds__(128, 1)
qkv_rope(const float* __restrict__ x,
         const unsigned short* __restrict__ wpack,
         unsigned short* __restrict__ qbf,
         unsigned short* __restrict__ kbf,
         unsigned short* __restrict__ vT) {
  int wave = threadIdx.x >> 5;
  int lane = threadIdx.x & 31;
  int unit = blockIdx.x * 4 + wave;   // 0..3071
  int mat  = unit % 3;
  int tmp  = unit / 3;
  int tb   = tmp & 127;               // T/16
  int b    = tmp >> 7;                // batch
  int idx  = lane & 15, hl = lane >> 4;

  const float* xrow = x + (size_t)(b * TT + tb * 16 + idx) * NEMBD;
  const unsigned short* wp = wpack + (size_t)mat * (32 * 8 * 32 * 16);

  v8f acc[8];
#pragma unroll
  for (int n8 = 0; n8 < 8; ++n8)
#pragma unroll
    for (int i = 0; i < 8; ++i) acc[n8][i] = 0.0f;

  for (int kb = 0; kb < 32; ++kb) {
    // A fragment: row (lane&15), K contiguous at hl*16 (f32 -> bf16)
    float av[16];
    const float4* c0 = reinterpret_cast<const float4*>(xrow + kb * 32 + hl * 16);
    const float4* c1 =
        reinterpret_cast<const float4*>(xrow + kb * 32 + hl * 16 + 8);
    *reinterpret_cast<float4*>(av + 0)  = c0[0];
    *reinterpret_cast<float4*>(av + 4)  = c0[1];
    *reinterpret_cast<float4*>(av + 8)  = c1[0];
    *reinterpret_cast<float4*>(av + 12) = c1[1];
    Frag16 a;
#pragma unroll
    for (int e = 0; e < 16; ++e) a.h[e] = (__bf16)av[e];

#pragma unroll
    for (int n8 = 0; n8 < 8; ++n8) {
      Frag16 bfr;
      const uint4* bp = reinterpret_cast<const uint4*>(
          wp + ((size_t)(kb * 8 + n8) * 32 + lane) * 16);
      bfr.q[0] = bp[0];
      bfr.q[1] = bp[1];
      acc[n8] = __builtin_amdgcn_wmma_f32_16x16x32_bf16(
          false, a.v, false, bfr.v, (short)0, acc[n8], false, false);
    }
  }

  if (mat < 2) {
    // RoPE then store row-major bf16.  C layout: row = r + 8*hl, col = idx.
    unsigned short* dst = (mat == 0) ? qbf : kbf;
#pragma unroll
    for (int n8 = 0; n8 < 8; ++n8) {
      int d = n8 * 16 + idx;
      int p = d >> 1;
      // freq = 10000^(-2p/128) ; ln(10000) = 9.210340371976184
      float freq = __expf(-(float)(2 * p) * (9.210340371976184f / 128.0f));
#pragma unroll
      for (int r = 0; r < 8; ++r) {
        int   tg    = tb * 16 + r + 8 * hl;
        float val   = acc[n8][r];
        float other = __shfl_xor(val, 1, 32);  // partner dim (d parity == lane parity)
        float ang = (float)tg * freq;
        float ca = __cosf(ang), sa = __sinf(ang);
        float res = (d & 1) ? (other * sa + val * ca)   // x1' = x0*s + x1*c
                            : (val * ca - other * sa);  // x0' = x0*c - x1*s
        dst[((size_t)(b * TT + tg) << 7) + d] = f32_bf16_bits(res);
      }
    }
  } else {
    // V: store transposed [B][128][T] bf16 for contiguous P*V B-fragments.
#pragma unroll
    for (int n8 = 0; n8 < 8; ++n8) {
      int d = n8 * 16 + idx;
#pragma unroll
      for (int r = 0; r < 8; ++r) {
        int tg = tb * 16 + r + 8 * hl;
        vT[((size_t)(b * HS + d) << 11) + tg] = f32_bf16_bits(acc[n8][r]);
      }
    }
  }
}

// ---------------------------------------------------------------------------
// Kernel 2: causal flash attention.  One wave owns 16 query rows, streams
// 32-column K/V blocks.  QK^T and P*V via v_wmma_f32_16x16x32_bf16; online
// softmax with half-wave shfl reductions; P transposed C->A layout through a
// 1KB-per-wave LDS bounce.
// ---------------------------------------------------------------------------
__global__ void __launch_bounds__(128, 1)
attn(const unsigned short* __restrict__ qbf,
     const unsigned short* __restrict__ kbf,
     const unsigned short* __restrict__ vT,
     float* __restrict__ out) {
  __shared__ __align__(16) unsigned short psmem[4 * 512];  // 16x32 bf16 / wave
  int wave = threadIdx.x >> 5;
  int lane = threadIdx.x & 31;
  int idx  = lane & 15, hl = lane >> 4;
  int unit = blockIdx.x * 4 + wave;  // 0..1023
  int qt   = unit & 127;
  int b    = unit >> 7;
  const float scale = 0.08838834764831845f;  // 1/sqrt(128)

  unsigned short* pl = psmem + wave * 512;

  // Q A-fragments for all 4 K-dim blocks of the head dim (row = lane&15)
  Frag16 qa[4];
  const unsigned short* qrow = qbf + ((size_t)(b * TT + qt * 16 + idx) << 7);
#pragma unroll
  for (int kb = 0; kb < 4; ++kb) {
    qa[kb].q[0] =
        *reinterpret_cast<const uint4*>(qrow + kb * 32 + hl * 16);
    qa[kb].q[1] =
        *reinterpret_cast<const uint4*>(qrow + kb * 32 + hl * 16 + 8);
  }

  float m_i[8], l_i[8];
#pragma unroll
  for (int r = 0; r < 8; ++r) { m_i[r] = -3.0e38f; l_i[r] = 0.0f; }
  v8f acc[8];
#pragma unroll
  for (int n8 = 0; n8 < 8; ++n8)
#pragma unroll
    for (int i = 0; i < 8; ++i) acc[n8][i] = 0.0f;

  int qmaxrow = qt * 16 + 15;
  for (int j0 = 0; j0 <= qmaxrow; j0 += 32) {
    if (j0 + 32 <= qmaxrow) {  // prefetch next K block (global_prefetch_b8)
      __builtin_prefetch(kbf + ((size_t)(b * TT + j0 + 32 + idx) << 7), 0, 0);
      __builtin_prefetch(kbf + ((size_t)(b * TT + j0 + 48 + idx) << 7), 0, 0);
    }
    // ---- S = Q * K^T over two 16-col tiles -------------------------------
    v8f S[2];
#pragma unroll
    for (int jh = 0; jh < 2; ++jh) {
#pragma unroll
      for (int i = 0; i < 8; ++i) S[jh][i] = 0.0f;
      const unsigned short* krow =
          kbf + ((size_t)(b * TT + j0 + jh * 16 + idx) << 7) + hl * 16;
#pragma unroll
      for (int kb = 0; kb < 4; ++kb) {
        Frag16 kf;
        const uint4* kp = reinterpret_cast<const uint4*>(krow + kb * 32);
        kf.q[0] = kp[0];
        kf.q[1] = kp[1];
        S[jh] = __builtin_amdgcn_wmma_f32_16x16x32_bf16(
            false, qa[kb].v, false, kf.v, (short)0, S[jh], false, false);
      }
    }
    // ---- scale + causal mask ---------------------------------------------
#pragma unroll
    for (int jh = 0; jh < 2; ++jh) {
      int col = j0 + jh * 16 + idx;
#pragma unroll
      for (int r = 0; r < 8; ++r) {
        int row = qt * 16 + r + 8 * hl;
        float s = S[jh][r] * scale;
        S[jh][r] = (col > row) ? -3.0e38f : s;
      }
    }
    // ---- online softmax (row stats via 16-lane half-wave reductions) -----
    float alpha[8];
#pragma unroll
    for (int r = 0; r < 8; ++r) {
      float v = fmaxf(S[0][r], S[1][r]);
#pragma unroll
      for (int off = 1; off < 16; off <<= 1) v = fmaxf(v, __shfl_xor(v, off, 32));
      float mn = fmaxf(m_i[r], v);
      alpha[r] = __expf(m_i[r] - mn);
      float p0 = __expf(S[0][r] - mn);
      float p1 = __expf(S[1][r] - mn);
      S[0][r] = p0;
      S[1][r] = p1;
      float ps = p0 + p1;
#pragma unroll
      for (int off = 1; off < 16; off <<= 1) ps += __shfl_xor(ps, off, 32);
      l_i[r] = l_i[r] * alpha[r] + ps;
      m_i[r] = mn;
    }
#pragma unroll
    for (int n8 = 0; n8 < 8; ++n8)
#pragma unroll
      for (int r = 0; r < 8; ++r) acc[n8][r] *= alpha[r];

    // ---- transpose P (C layout -> A layout) through LDS ------------------
#pragma unroll
    for (int jh = 0; jh < 2; ++jh)
#pragma unroll
      for (int r = 0; r < 8; ++r) {
        int m = r + 8 * hl;
        pl[m * 32 + jh * 16 + idx] = f32_bf16_bits(S[jh][r]);
      }
    Frag16 pf;  // A 16x32: row = idx, K contiguous at hl*16
    pf.q[0] = *reinterpret_cast<const uint4*>(pl + idx * 32 + hl * 16);
    pf.q[1] = *reinterpret_cast<const uint4*>(pl + idx * 32 + hl * 16 + 8);

    // ---- acc += P * V ----------------------------------------------------
#pragma unroll
    for (int n8 = 0; n8 < 8; ++n8) {
      Frag16 vf;  // B 32x16: col = head dim, K = j rows (contiguous in vT)
      const uint4* vp = reinterpret_cast<const uint4*>(
          vT + ((size_t)(b * HS + n8 * 16 + idx) << 11) + j0 + hl * 16);
      vf.q[0] = vp[0];
      vf.q[1] = vp[1];
      acc[n8] = __builtin_amdgcn_wmma_f32_16x16x32_bf16(
          false, pf.v, false, vf.v, (short)0, acc[n8], false, false);
    }
  }

  // ---- normalize + store (f32 output) ------------------------------------
  float invl[8];
#pragma unroll
  for (int r = 0; r < 8; ++r) invl[r] = 1.0f / l_i[r];
#pragma unroll
  for (int n8 = 0; n8 < 8; ++n8)
#pragma unroll
    for (int r = 0; r < 8; ++r) {
      int tg = qt * 16 + r + 8 * hl;
      out[((size_t)(b * TT + tg) << 7) + n8 * 16 + idx] = acc[n8][r] * invl[r];
    }
}

// ---------------------------------------------------------------------------
extern "C" void kernel_launch(void* const* d_in, const int* in_sizes, int n_in,
                              void* d_out, int out_size, void* d_ws,
                              size_t ws_size, hipStream_t stream) {
  const float* x  = (const float*)d_in[0];
  const float* Wq = (const float*)d_in[1];
  const float* Wk = (const float*)d_in[2];
  const float* Wv = (const float*)d_in[3];
  float* out = (float*)d_out;

  char* ws = (char*)d_ws;
  unsigned short* qbf   = (unsigned short*)(ws);                           // 4 MB
  unsigned short* kbf   = (unsigned short*)(ws + (size_t)4 * 1024 * 1024); // 4 MB
  unsigned short* vT    = (unsigned short*)(ws + (size_t)8 * 1024 * 1024); // 4 MB
  unsigned short* wpack = (unsigned short*)(ws + (size_t)12 * 1024 * 1024);// 768 KB

  // 3 * 32 * 8 * 32 = 24576 pack threads
  pack_weights<<<192, 128, 0, stream>>>(Wq, Wk, Wv, wpack);
  // B * T/16 * 3 = 3072 waves, 4 waves/block
  qkv_rope<<<768, 128, 0, stream>>>(x, wpack, qbf, kbf, vT);
  // B * T/16 = 1024 waves, 4 waves/block
  attn<<<256, 128, 0, stream>>>(qbf, kbf, vT, out);
}
